// LSTMDecoder_3504693314119
// MI455X (gfx1250) — matrix operand order repaired and checked by reading, not compiled, not run on hardware
//
#include <hip/hip_runtime.h>
#include <hip/hip_bf16.h>
#include <math.h>

// ---------------- problem constants ----------------
#define BB 32
#define EE 512
#define HH 1024
#define LL 1024
#define TT 256
#define H2 2048
#define H4 4096
#define PEE_BS ((TT + 1) * LL)   // 263168, per-batch stride of prev_e_enc_new
#define PHD_BS ((TT + 1) * HH)   // 263168

// output offsets (floats), concat in reference return order
#define OFF_EMB ((size_t)0)
#define OFF_H   (OFF_EMB + (size_t)BB * EE)
#define OFF_C   (OFF_H   + (size_t)BB * HH)
#define OFF_PEE (OFF_C   + (size_t)BB * HH)
#define OFF_PHD (OFF_PEE + (size_t)BB * PEE_BS)
#define OFF_CI  (OFF_PHD + (size_t)BB * PHD_BS)

// workspace offsets (floats)
#define WS_GATES   ((size_t)0)                       // 32*4096
#define WS_HRAW    (WS_GATES + (size_t)BB * H4)      // 32*1024
#define WS_HWE     (WS_HRAW  + (size_t)BB * HH)      // 32*2048
#define WS_HWD     (WS_HWE   + (size_t)BB * H2)      // 32*1024
#define WS_PREVMAX (WS_HWD   + (size_t)BB * HH)      // 32*1024
#define WS_PREVSUM (WS_PREVMAX + (size_t)BB * LL)    // 32*1024
#define WS_CTXACC  (WS_PREVSUM + (size_t)BB * LL)    // 32*2048
#define WS_SUMTMP  (WS_CTXACC + (size_t)BB * H2)     // 32
#define WS_EDEC    (WS_SUMTMP + (size_t)BB)          // 32*256
#define WS_ATDEC   (WS_EDEC   + (size_t)BB * TT)     // 32*256
#define WS_BSUM    (WS_ATDEC  + (size_t)BB * TT)     // 4096

typedef __attribute__((ext_vector_type(16))) _Float16 v16h;
typedef __attribute__((ext_vector_type(8)))  float    v8f;

#define NEGINF (-__builtin_inff())

// -------- wave32 cross-lane reductions via ds_swizzle (xor patterns) --------
#define SWZ(x, imm) __int_as_float(__builtin_amdgcn_ds_swizzle(__float_as_int(x), imm))
__device__ __forceinline__ float wred_add(float x) {
  x += SWZ(x, 0x041f);  // xor 1
  x += SWZ(x, 0x081f);  // xor 2
  x += SWZ(x, 0x101f);  // xor 4
  x += SWZ(x, 0x201f);  // xor 8
  x += SWZ(x, 0x401f);  // xor 16
  return x;
}
__device__ __forceinline__ float wred_max(float x) {
  x = fmaxf(x, SWZ(x, 0x041f));
  x = fmaxf(x, SWZ(x, 0x081f));
  x = fmaxf(x, SWZ(x, 0x101f));
  x = fmaxf(x, SWZ(x, 0x201f));
  x = fmaxf(x, SWZ(x, 0x401f));
  return x;
}

__device__ __forceinline__ float sigf(float x) { return 1.0f / (1.0f + __expf(-x)); }

// =====================================================================
// WMMA GEMM:  D[M,N] = A1[M,K1] * op(W1) (+ A2[M,K2] * op(W2)) (+ bias)
//   op(W) = W^T if TB (W stored [N,K] row-major) else W (stored [K,N]).
//   TB / HASA2 / HASBIAS are template params -> branch-free, vectorizable
//   inner loops (TB=1: 16 consecutive floats per lane per step).
//   grid = (N/16, M/16), block = 32 (one wave per 16x16 D tile).
//   f32 C/D layout per CDNA5 ISA 7.12.2; f16 A/B use contiguous k = h + 16*hi (same order in both).
// =====================================================================
template <int TB>
__device__ __forceinline__ v8f gemm_acc(v8f acc, const float* __restrict__ A, int lda,
                                        const float* __restrict__ W, int ldw, int K,
                                        int m0, int n0, int lane) {
  const int hi = lane >> 4;       // 0/1
  const int lo = lane & 15;
  const float* __restrict__ arow = A + (size_t)(m0 + lo) * lda;       // lane's A row
  const float* __restrict__ wb =
      TB ? (W + (size_t)(n0 + lo) * ldw)   // row n of W[N,K] == column n of B
         : (W + (n0 + lo));                // column n of W[K,N]
  for (int kk = 0; kk < K; kk += 32) {
    v16h a, b;
#pragma unroll
    for (int h = 0; h < 16; ++h) {
      // A operand: lane gives row, this code's contiguous order K = h + 16*hi (same as B)
      int ak = kk + h + 16 * hi;
      a[h] = (_Float16)arow[ak];
      // B operand: lane gives column, this code's contiguous order K = h + 16*hi (same as A)
      int bk = kk + h + 16 * hi;
      b[h] = (_Float16)(TB ? wb[bk] : wb[(size_t)bk * ldw]);
    }
    acc = __builtin_amdgcn_wmma_f32_16x16x32_f16(false, a, false, b, (short)0, acc,
                                                 false, false);
  }
  return acc;
}

template <int TB1, int TB2, bool HASA2, bool HASBIAS>
__global__ void gemm_wmma(const float* __restrict__ A1, int lda1,
                          const float* __restrict__ W1, int ldw1, int K1,
                          const float* __restrict__ A2, int lda2,
                          const float* __restrict__ W2, int ldw2, int K2,
                          const float* __restrict__ bias,
                          float* __restrict__ D, int ldd) {
  const int lane = threadIdx.x;
  const int hi = lane >> 4, lo = lane & 15;
  const int n0 = blockIdx.x * 16;
  const int m0 = blockIdx.y * 16;

  v8f acc;
  float bv = HASBIAS ? bias[n0 + lo] : 0.0f;
#pragma unroll
  for (int r = 0; r < 8; ++r) acc[r] = bv;

  acc = gemm_acc<TB1>(acc, A1, lda1, W1, ldw1, K1, m0, n0, lane);
  if (HASA2)
    acc = gemm_acc<TB2>(acc, A2, lda2, W2, ldw2, K2, m0, n0, lane);

  // C/D 16x16 f32 layout: VGPR r -> row m0 + r + 8*hi, col n0 + lo
#pragma unroll
  for (int r = 0; r < 8; ++r)
    D[(size_t)(m0 + r + 8 * hi) * ldd + (n0 + lo)] = acc[r];
}

// =====================================================================
// init: zero atomic accumulators, bsum = b_ih + b_hh
// =====================================================================
__global__ void k_init(float* __restrict__ ws, const float* __restrict__ b_ih,
                       const float* __restrict__ b_hh) {
  int i = blockIdx.x * blockDim.x + threadIdx.x;      // 65536 threads
  ws[WS_CTXACC + i] = 0.0f;
  if (i < BB) ws[WS_SUMTMP + i] = 0.0f;
  if (i < H4) ws[WS_BSUM + i] = b_ih[i] + b_hh[i];
}

// =====================================================================
// LSTM elementwise: gates[B,4H] + c0 -> c (out), h_raw (ws)
// =====================================================================
__global__ void k_lstm_ew(const float* __restrict__ gates,
                          const float* __restrict__ c0,
                          float* __restrict__ out_c, float* __restrict__ hraw) {
  int i = blockIdx.x * blockDim.x + threadIdx.x;      // 32768 threads
  int b = i >> 10, j = i & (HH - 1);
  const float* g = gates + (size_t)b * H4;
  float gi = g[j], gf = g[HH + j], gg = g[2 * HH + j], go = g[3 * HH + j];
  float cn = sigf(gf) * c0[i] + sigf(gi) * tanhf(gg);
  out_c[i] = cn;
  hraw[i] = sigf(go) * tanhf(cn);
}

// =====================================================================
// LayerNorm over H per row -> h written to: out_h, context_info[:, :H],
// prev_h_dec_new row T.   grid = 32 blocks (one per b), 256 threads.
// =====================================================================
__global__ void k_ln(const float* __restrict__ hraw, const float* __restrict__ gamma,
                     const float* __restrict__ beta, float* __restrict__ out) {
  __shared__ float red1[8], red2[8];
  int b = blockIdx.x, tid = threadIdx.x;
  const float* x = hraw + (size_t)b * HH;
  float s1 = 0.f, s2 = 0.f;
#pragma unroll
  for (int k = 0; k < 4; ++k) {
    float v = x[k * 256 + tid];
    s1 += v; s2 += v * v;
  }
  s1 = wred_add(s1); s2 = wred_add(s2);
  if ((tid & 31) == 0) { red1[tid >> 5] = s1; red2[tid >> 5] = s2; }
  __syncthreads();
  float t1 = 0.f, t2 = 0.f;
#pragma unroll
  for (int w = 0; w < 8; ++w) { t1 += red1[w]; t2 += red2[w]; }
  float mu = t1 / HH;
  float var = t2 / HH - mu * mu;
  float rs = rsqrtf(var + 1e-5f);
#pragma unroll
  for (int k = 0; k < 4; ++k) {
    int j = k * 256 + tid;
    float h = (x[j] - mu) * rs * gamma[j] + beta[j];
    out[OFF_H + (size_t)b * HH + j] = h;
    out[OFF_CI + (size_t)b * H4 + j] = h;
    out[OFF_PHD + (size_t)b * PHD_BS + (size_t)TT * HH + j] = h;
  }
}

// =====================================================================
// prev_e_enc pass: copy to output + online softmax stats over T per (b,l).
// grid = (4, 32) : blockIdx.y=b, 256 threads -> l = bx*256+tid. One 33MB pass.
// =====================================================================
__global__ void k_enc_stats(const float* __restrict__ pee, float* __restrict__ out,
                            float* __restrict__ ws) {
  int b = blockIdx.y;
  int l = blockIdx.x * 256 + threadIdx.x;
  const float* src = pee + (size_t)b * TT * LL + l;
  float* dst = out + OFF_PEE + (size_t)b * PEE_BS + l;
  float mx = NEGINF, s = 0.0f;
  for (int t = 0; t < TT; ++t) {
    float v = src[(size_t)t * LL];
    dst[(size_t)t * LL] = v;
    if (v > mx) { s *= __expf(mx - v); mx = v; }
    s += __expf(v - mx);
  }
  ws[WS_PREVMAX + (size_t)b * LL + l] = mx;
  ws[WS_PREVSUM + (size_t)b * LL + l] = s;
}

// =====================================================================
// Single fused pass over enc_out (the 268MB tensor):
//   e = <hw_e[b], enc_out[b,l,:]>;  m = max(e_masked, prevmax);
//   tmp = exp(e-m) / (prevsum*exp(prevmax-m) + eps);
//   ctxacc[b,:] += tmp * enc_out[b,l,:] (atomics); sumtmp[b] += tmp;
//   also appends masked e_enc as row T of prev_e_enc_new.
// grid = (8, 32): blockIdx.y=b, 128 rows per block, 256 threads.
// =====================================================================
__global__ void k_enc_fused(const float* __restrict__ enc_out,
                            const unsigned char* __restrict__ enc_mask,
                            float* __restrict__ ws, float* __restrict__ out) {
  __shared__ float hwe_s[H2];
  __shared__ float red[8];
  int b = blockIdx.y, tid = threadIdx.x;
  for (int j = tid; j < H2; j += 256) hwe_s[j] = ws[WS_HWE + (size_t)b * H2 + j];
  __syncthreads();

  float ctx[8];
#pragma unroll
  for (int k = 0; k < 8; ++k) ctx[k] = 0.0f;
  float stmp = 0.0f;

  int l0 = blockIdx.x * 128;
  for (int r = 0; r < 128; ++r) {
    int l = l0 + r;
    const float* row = enc_out + ((size_t)b * LL + l) * H2;
    float v[8];
    float dp = 0.0f;
#pragma unroll
    for (int k = 0; k < 8; ++k) {
      int j = k * 256 + tid;
      v[k] = row[j];
      dp += v[k] * hwe_s[j];
    }
    dp = wred_add(dp);
    if ((tid & 31) == 0) red[tid >> 5] = dp;
    __syncthreads();
    float e = 0.0f;
#pragma unroll
    for (int w = 0; w < 8; ++w) e += red[w];
    __syncthreads();

    bool msk = enc_mask[(size_t)b * LL + l] != 0;
    float pm = ws[WS_PREVMAX + (size_t)b * LL + l];
    float ps = ws[WS_PREVSUM + (size_t)b * LL + l];
    float ee = msk ? e : NEGINF;
    float m = fmaxf(ee, pm);
    m = msk ? m : 0.0f;
    float num = __expf(ee - m);
    float den = ps * __expf(pm - m) + 0.001f;
    float tmp = num / den;
    if (tid == 0) {
      stmp += tmp;
      out[OFF_PEE + (size_t)b * PEE_BS + (size_t)TT * LL + l] = ee;
    }
#pragma unroll
    for (int k = 0; k < 8; ++k) ctx[k] += tmp * v[k];
  }

#pragma unroll
  for (int k = 0; k < 8; ++k)
    atomicAdd(&ws[WS_CTXACC + (size_t)b * H2 + k * 256 + tid], ctx[k]);
  if (tid == 0) atomicAdd(&ws[WS_SUMTMP + b], stmp);
}

// context_enc = ctxacc / (sumtmp + eps)  ->  context_info[:, H:3H]
__global__ void k_enc_fin(const float* __restrict__ ws, float* __restrict__ out) {
  int i = blockIdx.x * blockDim.x + threadIdx.x;   // 65536
  int b = i >> 11, j = i & (H2 - 1);
  out[OFF_CI + (size_t)b * H4 + HH + j] =
      ws[WS_CTXACC + i] / (ws[WS_SUMTMP + b] + 0.001f);
}

// =====================================================================
// decoder attention pass 1: e_dec[b,t] = <hw_d[b], prev_h_dec[b,t,:]>
// fused with the copy of prev_h_dec into the output.
// grid = (8, 32): 32 t-rows per block, 256 threads.
// =====================================================================
__global__ void k_dec_scores(const float* __restrict__ phd, float* __restrict__ ws,
                             float* __restrict__ out) {
  __shared__ float hwd_s[HH];
  __shared__ float red[8];
  int b = blockIdx.y, tid = threadIdx.x;
  for (int j = tid; j < HH; j += 256) hwd_s[j] = ws[WS_HWD + (size_t)b * HH + j];
  __syncthreads();
  int t0 = blockIdx.x * 32;
  for (int r = 0; r < 32; ++r) {
    int t = t0 + r;
    const float* row = phd + ((size_t)b * TT + t) * HH;
    float* dst = out + OFF_PHD + (size_t)b * PHD_BS + (size_t)t * HH;
    float dp = 0.0f;
#pragma unroll
    for (int k = 0; k < 4; ++k) {
      int j = k * 256 + tid;
      float v = row[j];
      dst[j] = v;
      dp += v * hwd_s[j];
    }
    dp = wred_add(dp);
    if ((tid & 31) == 0) red[tid >> 5] = dp;
    __syncthreads();
    if (tid == 0) {
      float e = 0.0f;
#pragma unroll
      for (int w = 0; w < 8; ++w) e += red[w];
      ws[WS_EDEC + (size_t)b * TT + t] = e;
    }
    __syncthreads();
  }
}

// decoder softmax over T (masked).  grid = 32, 256 threads (t = tid).
__global__ void k_dec_softmax(const unsigned char* __restrict__ dec_mask,
                              float* __restrict__ ws) {
  __shared__ float red[8];
  int b = blockIdx.x, t = threadIdx.x;
  bool msk = dec_mask[(size_t)b * TT + t] != 0;
  float e = ws[WS_EDEC + (size_t)b * TT + t];
  float ee = msk ? e : NEGINF;
  float mx = wred_max(ee);
  if ((t & 31) == 0) red[t >> 5] = mx;
  __syncthreads();
  mx = NEGINF;
#pragma unroll
  for (int w = 0; w < 8; ++w) mx = fmaxf(mx, red[w]);
  __syncthreads();
  float p = __expf(ee - (msk ? mx : 0.0f));
  float s = wred_add(p);
  if ((t & 31) == 0) red[t >> 5] = s;
  __syncthreads();
  s = 0.0f;
#pragma unroll
  for (int w = 0; w < 8; ++w) s += red[w];
  ws[WS_ATDEC + (size_t)b * TT + t] = p / s;
}

// context_dec[b,j] = sum_t at_dec[b,t]*prev_h_dec[b,t,j] -> context_info[:,3H:4H]
// grid = (4, 32): 256 threads, one j each. (2nd 33MB read; L2-resident.)
__global__ void k_dec_ctx(const float* __restrict__ phd, const float* __restrict__ ws,
                          float* __restrict__ out) {
  __shared__ float at_s[TT];
  int b = blockIdx.y, tid = threadIdx.x;
  at_s[tid] = ws[WS_ATDEC + (size_t)b * TT + tid];
  __syncthreads();
  int j = blockIdx.x * 256 + tid;
  const float* base = phd + (size_t)b * TT * HH + j;
  float c = 0.0f;
  for (int t = 0; t < TT; ++t) c += at_s[t] * base[(size_t)t * HH];
  out[OFF_CI + (size_t)b * H4 + 3 * HH + j] = c;
}

// =====================================================================
extern "C" void kernel_launch(void* const* d_in, const int* in_sizes, int n_in,
                              void* d_out, int out_size, void* d_ws, size_t ws_size,
                              hipStream_t stream) {
  const float* x_t   = (const float*)d_in[0];
  const float* h0    = (const float*)d_in[1];
  const float* c0    = (const float*)d_in[2];
  const float* enc_out = (const float*)d_in[3];
  const float* pee   = (const float*)d_in[4];
  const float* phd   = (const float*)d_in[5];
  const float* W_ih  = (const float*)d_in[6];
  const float* W_hh  = (const float*)d_in[7];
  const float* b_ih  = (const float*)d_in[8];
  const float* b_hh  = (const float*)d_in[9];
  const float* ln_g  = (const float*)d_in[10];
  const float* ln_b  = (const float*)d_in[11];
  const float* W_enc = (const float*)d_in[12];
  const float* W_dec = (const float*)d_in[13];
  const float* W_gen = (const float*)d_in[14];
  const float* b_gen = (const float*)d_in[15];
  const unsigned char* enc_mask = (const unsigned char*)d_in[16];
  const unsigned char* dec_mask = (const unsigned char*)d_in[17];

  float* out = (float*)d_out;
  float* ws  = (float*)d_ws;

  // 1) zero accumulators, bsum = b_ih + b_hh
  k_init<<<256, 256, 0, stream>>>(ws, b_ih, b_hh);

  // 2) gates = x@W_ih^T + h0@W_hh^T + bsum   [32,4096]  (WMMA, both B transposed)
  gemm_wmma<1, 1, true, true><<<dim3(H4 / 16, 2), 32, 0, stream>>>(
      x_t, EE, W_ih, EE, EE, h0, HH, W_hh, HH, HH, ws + WS_BSUM,
      ws + WS_GATES, H4);

  // 3) LSTM elementwise -> c (out), h_raw (ws)
  k_lstm_ew<<<128, 256, 0, stream>>>(ws + WS_GATES, c0, out + OFF_C, ws + WS_HRAW);

  // 4) LayerNorm -> h (out_h, context_info[:, :H], prev_h_dec_new row T)
  k_ln<<<BB, 256, 0, stream>>>(ws + WS_HRAW, ln_g, ln_b, out);

  // 5) hw_e = h @ W_enc_att [32,2048]; 6) hw_d = h @ W_dec_att [32,1024]  (WMMA)
  gemm_wmma<0, 0, false, false><<<dim3(H2 / 16, 2), 32, 0, stream>>>(
      out + OFF_H, HH, W_enc, H2, HH, nullptr, 0, nullptr, 0, 0, nullptr,
      ws + WS_HWE, H2);
  gemm_wmma<0, 0, false, false><<<dim3(HH / 16, 2), 32, 0, stream>>>(
      out + OFF_H, HH, W_dec, HH, HH, nullptr, 0, nullptr, 0, 0, nullptr,
      ws + WS_HWD, HH);

  // 7) prev_e_enc: online stats over T + copy to output (single 33MB pass)
  k_enc_stats<<<dim3(4, BB), 256, 0, stream>>>(pee, out, ws);

  // 8) fused encoder attention: single 268MB pass over enc_out
  k_enc_fused<<<dim3(8, BB), 256, 0, stream>>>(enc_out, enc_mask, ws, out);

  // 9) normalize encoder context -> context_info[:, H:3H]
  k_enc_fin<<<256, 256, 0, stream>>>(ws, out);

  // 10) decoder scores + copy prev_h_dec to output
  k_dec_scores<<<dim3(8, BB), 256, 0, stream>>>(phd, ws, out);

  // 11) decoder softmax
  k_dec_softmax<<<BB, 256, 0, stream>>>(dec_mask, ws);

  // 12) decoder context -> context_info[:, 3H:4H]
  k_dec_ctx<<<dim3(4, BB), 256, 0, stream>>>(phd, ws, out);

  // 13) embeddings = context_info @ W_gen^T + b_gen   [32,512]  (WMMA)
  gemm_wmma<1, 0, false, true><<<dim3(EE / 16, 2), 32, 0, stream>>>(
      out + OFF_CI, H4, W_gen, H4, H4, nullptr, 0, nullptr, 0, 0, b_gen,
      out + OFF_EMB, EE);
}
